// LinearAttentionBlock_58317065945866
// MI455X (gfx1250) — matrix operand order repaired and checked by reading, not compiled, not run on hardware
//
#include <hip/hip_runtime.h>
#include <hip/hip_bf16.h>

// ---------------- problem constants ----------------
#define Bz   4
#define Tz   2048
#define FIN  512
#define Dz   128
#define BT   (Bz * Tz)          // 8192
#define TC   128                // chunk length for the S-scan
#define NC   (Tz / TC)          // 16 chunks per batch
#define CLAMP_V 1e20f
#define EPS_LN  1e-5f
#define EPS_DEN 1e-5f

typedef _Float16 hfloat;
typedef __attribute__((ext_vector_type(16))) _Float16 v16h;
typedef __attribute__((ext_vector_type(8)))  _Float16 v8h;
typedef __attribute__((ext_vector_type(8)))  float    v8f;

// ---------------- WMMA tile loaders (wave32, per CDNA5 ISA layouts) ----------
// A (16x32 f16, MxK): lanes 0-15 hold M=0..15 / K{0..7,16..23}; lanes 16-31 hold
// M=0..15 / K{8..15,24..31}. Rows are contiguous in memory (A row-major, lda=K).
__device__ inline v16h load_a_tile(const hfloat* A, int lda, int m0, int k0, int lane) {
    int m   = lane & 15;
    int off = (lane < 16) ? 0 : 8;
    const hfloat* p = A + (size_t)(m0 + m) * lda + k0 + off;
    v8h lo = *(const v8h*)p;          // K = off .. off+7
    v8h hi = *(const v8h*)(p + 16);   // K = off+16 .. off+23
    v16h r;
    #pragma unroll
    for (int i = 0; i < 8; ++i) { r[i] = lo[i]; r[i + 8] = hi[i]; }
    return r;
}

// B (32x16 f16, KxN) supplied pre-transposed Bt[N][K] row-major: lane n=lane&15
// holds K{0..7,16..23} (lanes 0-15) or K{8..15,24..31} (lanes 16-31), matching A.
__device__ inline v16h load_bt_tile(const hfloat* Bt, int ldb, int n0, int k0, int lane) {
    int n   = lane & 15;
    int off = (lane < 16) ? 0 : 8;
    const hfloat* p = Bt + (size_t)(n0 + n) * ldb + k0 + off;
    v8h lo = *(const v8h*)p;
    v8h hi = *(const v8h*)(p + 16);
    v16h r;
    #pragma unroll
    for (int i = 0; i < 8; ++i) { r[i] = lo[i]; r[i + 8] = hi[i]; }
    return r;
}

// ---------------- kernels ----------------

// LayerNorm over FIN, write f16 normalized x
__global__ __launch_bounds__(256) void ln_kernel(const float* __restrict__ hist,
                                                 const float* __restrict__ g,
                                                 const float* __restrict__ bta,
                                                 hfloat* __restrict__ Xh) {
    __shared__ float s1[256], s2[256];
    int row = blockIdx.x, tid = threadIdx.x;
    const float* hr = hist + (size_t)row * FIN;
    float a0 = hr[tid], a1 = hr[tid + 256];
    s1[tid] = a0 + a1;
    s2[tid] = a0 * a0 + a1 * a1;
    __syncthreads();
    for (int s = 128; s > 0; s >>= 1) {
        if (tid < s) { s1[tid] += s1[tid + s]; s2[tid] += s2[tid + s]; }
        __syncthreads();
    }
    float mu  = s1[0] * (1.0f / FIN);
    float var = s2[0] * (1.0f / FIN) - mu * mu;
    float inv = rsqrtf(var + EPS_LN);
    hfloat* xr = Xh + (size_t)row * FIN;
    xr[tid]       = (hfloat)(((a0 - mu) * inv) * g[tid]       + bta[tid]);
    xr[tid + 256] = (hfloat)(((a1 - mu) * inv) * g[tid + 256] + bta[tid + 256]);
}

// Convert + transpose weights to f16: WtP[n][k] for concatenated [Wk|Wq|Wv|sc_w],
// W1t[n][k]=w1[k][n], W2t[n][k]=w2[k][n].
__global__ void wcvt_kernel(const float* __restrict__ Wk, const float* __restrict__ Wq,
                            const float* __restrict__ Wv, const float* __restrict__ scw,
                            const float* __restrict__ w1, const float* __restrict__ w2,
                            hfloat* __restrict__ WtP, hfloat* __restrict__ W1t,
                            hfloat* __restrict__ W2t) {
    int idx = blockIdx.x * blockDim.x + threadIdx.x;
    if (idx < 512 * 512) {
        int n = idx / 512, k = idx % 512;
        const float* src = (n < 128) ? Wk : (n < 256) ? Wq : (n < 384) ? Wv : scw;
        WtP[(size_t)n * 512 + k] = (hfloat)src[(size_t)k * Dz + (n & 127)];
    } else {
        int r = idx - 512 * 512;
        if (r < Dz * Dz) {
            int n = r / Dz, k = r % Dz;
            W1t[r] = (hfloat)w1[(size_t)k * Dz + n];
        } else {
            int r2 = r - Dz * Dz;
            int n = r2 / Dz, k = r2 % Dz;
            W2t[r2] = (hfloat)w2[(size_t)k * Dz + n];
        }
    }
}

// Generic WMMA GEMM: C[MxN] = A[MxK] (f16 rowmajor) * Bt[NxK]^T, f32 accumulate.
__global__ __launch_bounds__(32) void gemm_wmma_kernel(const hfloat* __restrict__ A,
                                                       const hfloat* __restrict__ Bt,
                                                       float* __restrict__ C,
                                                       int N, int KK) {
    int lane = threadIdx.x;
    int n0 = blockIdx.x * 16;
    int m0 = blockIdx.y * 16;
    v8f acc = {};
    for (int k0 = 0; k0 < KK; k0 += 32) {
        v16h a = load_a_tile(A, KK, m0, k0, lane);
        v16h b = load_bt_tile(Bt, KK, n0, k0, lane);
        acc = __builtin_amdgcn_wmma_f32_16x16x32_f16(false, a, false, b,
                                                     (short)0, acc, false, false);
    }
    int n     = n0 + (lane & 15);
    int mbase = m0 + ((lane >> 4) << 3);   // C layout: VGPR r -> M=r (+8 for hi lanes)
    #pragma unroll
    for (int r = 0; r < 8; ++r)
        C[(size_t)(mbase + r) * N + n] = acc[r];
}

// Split fused projection output, apply phi=elu+1 to K,Q; f16 mirrors of K,V.
__global__ void split_phi_kernel(const float* __restrict__ PRJ,
                                 float* __restrict__ Kf, float* __restrict__ Qf,
                                 float* __restrict__ Vf, float* __restrict__ XSf,
                                 hfloat* __restrict__ Kh, hfloat* __restrict__ Vh) {
    size_t idx = (size_t)blockIdx.x * blockDim.x + threadIdx.x;
    if (idx >= (size_t)BT * 512) return;
    int n = (int)(idx & 511);
    size_t row = idx >> 9;
    float v = PRJ[idx];
    size_t o = row * Dz + (n & 127);
    if (n < 128)       { float p = v > 0.f ? v + 1.0f : __expf(v); Kf[o] = p; Kh[o] = (hfloat)p; }
    else if (n < 256)  { Qf[o] = v > 0.f ? v + 1.0f : __expf(v); }
    else if (n < 384)  { Vf[o] = v; Vh[o] = (hfloat)v; }
    else               { XSf[o] = v; }
}

// Z = clip(cumsum_t K) + Z0, written straight to output region 2.
__global__ void zcum_kernel(const float* __restrict__ Kf, const float* __restrict__ Z0,
                            float* __restrict__ Zout) {
    int tid = blockIdx.x * blockDim.x + threadIdx.x;  // B*D = 512
    if (tid >= Bz * Dz) return;
    int b = tid >> 7, d = tid & 127;
    float acc = 0.f, z0 = Z0[tid];
    for (int t = 0; t < Tz; ++t) {
        size_t o = ((size_t)b * Tz + t) * Dz + d;
        acc += Kf[o];
        Zout[o] = fminf(fmaxf(acc, -CLAMP_V), CLAMP_V) + z0;
    }
}

// Per-chunk KV sums: CS[b,c] = Kc^T @ Vc  (128x128x128 GEMM via WMMA).
__global__ __launch_bounds__(32) void chunksum_kernel(const hfloat* __restrict__ Kh,
                                                      const hfloat* __restrict__ Vh,
                                                      float* __restrict__ CS) {
    int lane = threadIdx.x;
    int j0 = blockIdx.x * 16;   // N tile
    int i0 = blockIdx.y * 16;   // M tile
    int bc = blockIdx.z;        // b*NC + c
    const hfloat* Kc = Kh + (size_t)bc * TC * Dz;
    const hfloat* Vc = Vh + (size_t)bc * TC * Dz;
    int m = lane & 15, hi = lane >> 4;
    v8f acc = {};
    for (int k0 = 0; k0 < TC; k0 += 32) {
        v16h a = {}, b = {};
        #pragma unroll
        for (int v = 0; v < 8; ++v) {
            int ka = 2 * v + hi * 16;  // A layout K index
            a[2 * v]     = Kc[(size_t)(k0 + ka)     * Dz + i0 + m];  // A = K^T
            a[2 * v + 1] = Kc[(size_t)(k0 + ka + 1) * Dz + i0 + m];
            int kb = 2 * v + hi * 16;                                // B layout K index
            b[2 * v]     = Vc[(size_t)(k0 + kb)     * Dz + j0 + m];
            b[2 * v + 1] = Vc[(size_t)(k0 + kb + 1) * Dz + j0 + m];
        }
        acc = __builtin_amdgcn_wmma_f32_16x16x32_f16(false, a, false, b,
                                                     (short)0, acc, false, false);
    }
    int n = j0 + m, mb = i0 + hi * 8;
    #pragma unroll
    for (int r = 0; r < 8; ++r)
        CS[(size_t)bc * Dz * Dz + (size_t)(mb + r) * Dz + n] = acc[r];
}

// Exclusive prefix over chunks (per batch, elementwise over DxD).
__global__ void prefix_kernel(const float* __restrict__ CS, float* __restrict__ Pfx) {
    int tid = blockIdx.x * blockDim.x + threadIdx.x;  // B*D*D = 65536
    int b = tid >> 14, ij = tid & 16383;
    float acc = 0.f;
    for (int c = 0; c < NC; ++c) {
        size_t o = ((size_t)(b * NC + c)) * (Dz * Dz) + ij;
        Pfx[o] = acc;
        acc += CS[o];
    }
}

// Fused scan: stream S_t = clip(prefix + running KV) + S0 to HBM (the 536 MB),
// and reduce numerator num[t,j] = sum_i Q[t,i]*S_t[i,j] via LDS.
__global__ __launch_bounds__(1024) void scan_kernel(const float* __restrict__ Kf,
                                                    const float* __restrict__ Vf,
                                                    const float* __restrict__ Qf,
                                                    const float* __restrict__ Pfx,
                                                    const float* __restrict__ S0,
                                                    float* __restrict__ Sout,
                                                    float* __restrict__ NUM) {
    __shared__ float lK[Dz], lV[Dz], lQ[Dz], red[1024];
    int b = blockIdx.x >> 4, c = blockIdx.x & 15;
    int tid = threadIdx.x;
    int j = tid & 127, ig = tid >> 7;   // ig in 0..7, rows i = ig + 8r
    float s[16];
    size_t base = ((size_t)(b * NC + c)) * (Dz * Dz);
    #pragma unroll
    for (int r = 0; r < 16; ++r) s[r] = Pfx[base + (size_t)(ig + 8 * r) * Dz + j];
    const float* S0b = S0 + (size_t)b * Dz * Dz;

    for (int tt = 0; tt < TC; ++tt) {
        size_t row = (size_t)b * Tz + c * TC + tt;
        if (tid < 128)       lK[tid]        = Kf[row * Dz + tid];
        else if (tid < 256)  lV[tid - 128]  = Vf[row * Dz + (tid - 128)];
        else if (tid < 384)  lQ[tid - 256]  = Qf[row * Dz + (tid - 256)];
        if (tid < 128 && tt + 1 < TC)       // warm L2/WGP$ for next timestep
            __builtin_prefetch(&Kf[(row + 1) * Dz + tid], 0, 3);
        __syncthreads();

        float v  = lV[j];
        float pn = 0.f;
        float* Srow = Sout + row * (Dz * Dz);
        #pragma unroll
        for (int r = 0; r < 16; ++r) {
            int i = ig + 8 * r;
            s[r] += lK[i] * v;
            float sc = fminf(fmaxf(s[r], -CLAMP_V), CLAMP_V) + S0b[(size_t)i * Dz + j];
            Srow[(size_t)i * Dz + j] = sc;      // coalesced 512B rows
            pn += lQ[i] * sc;
        }
        red[tid] = pn;
        __syncthreads();
        if (ig == 0) {
            float a = 0.f;
            #pragma unroll
            for (int g = 0; g < 8; ++g) a += red[j + 128 * g];
            NUM[row * Dz + j] = a;
        }
    }
}

// denominator + attention output in f16 for the FFN GEMMs
__global__ void den_attn_kernel(const float* __restrict__ Qf, const float* __restrict__ Zout,
                                const float* __restrict__ NUM, hfloat* __restrict__ Hh) {
    int row = blockIdx.x * blockDim.x + threadIdx.x;
    if (row >= BT) return;
    size_t o = (size_t)row * Dz;
    float den = EPS_DEN;
    for (int d = 0; d < Dz; ++d) den += Qf[o + d] * Zout[o + d];
    float inv = 1.0f / den;
    for (int d = 0; d < Dz; ++d) Hh[o + d] = (hfloat)(NUM[o + d] * inv);
}

__global__ void bias_relu_kernel(const float* __restrict__ F, const float* __restrict__ bias,
                                 hfloat* __restrict__ Gh) {
    size_t idx = (size_t)blockIdx.x * blockDim.x + threadIdx.x;
    if (idx >= (size_t)BT * Dz) return;
    float v = F[idx] + bias[idx & 127];
    Gh[idx] = (hfloat)(v > 0.f ? v : 0.f);
}

__global__ void final_kernel(const float* __restrict__ F2, const float* __restrict__ b2,
                             const float* __restrict__ XSf, const float* __restrict__ scb,
                             float* __restrict__ Out0) {
    size_t idx = (size_t)blockIdx.x * blockDim.x + threadIdx.x;
    if (idx >= (size_t)BT * Dz) return;
    int n = (int)(idx & 127);
    float v = F2[idx] + b2[n];
    v = v > 0.f ? v : 0.f;
    Out0[idx] = v + XSf[idx] + scb[n];
}

// ---------------- host-side orchestration ----------------
extern "C" void kernel_launch(void* const* d_in, const int* in_sizes, int n_in,
                              void* d_out, int out_size, void* d_ws, size_t ws_size,
                              hipStream_t stream) {
    const float* hist = (const float*)d_in[0];
    const float* S0   = (const float*)d_in[1];
    const float* Z0   = (const float*)d_in[2];
    const float* Wk   = (const float*)d_in[3];
    const float* Wq   = (const float*)d_in[4];
    const float* Wv   = (const float*)d_in[5];
    const float* ln_g = (const float*)d_in[6];
    const float* ln_b = (const float*)d_in[7];
    const float* w1   = (const float*)d_in[8];
    const float* b1   = (const float*)d_in[9];
    const float* w2   = (const float*)d_in[10];
    const float* b2   = (const float*)d_in[11];
    const float* sc_w = (const float*)d_in[12];
    const float* sc_b = (const float*)d_in[13];

    float* Out0 = (float*)d_out;                      // [B,T,D]
    float* Sout = Out0 + (size_t)BT * Dz;             // [B,T,D,D]
    float* Zout = Sout + (size_t)BT * Dz * Dz;        // [B,T,D]

    // workspace carve-out
    char* w = (char*)d_ws;
    auto alloc = [&](size_t bytes) { char* p = w; w += (bytes + 255) & ~(size_t)255; return p; };
    hfloat* Xh   = (hfloat*)alloc((size_t)BT * FIN * 2);       // 8 MB
    hfloat* WtP  = (hfloat*)alloc((size_t)512 * 512 * 2);
    hfloat* W1t  = (hfloat*)alloc((size_t)Dz * Dz * 2);
    hfloat* W2t  = (hfloat*)alloc((size_t)Dz * Dz * 2);
    float*  PRJ  = (float*) alloc((size_t)BT * 512 * 4);       // 16 MB, reused for F1/F2
    float*  Kf   = (float*) alloc((size_t)BT * Dz * 4);
    float*  Qf   = (float*) alloc((size_t)BT * Dz * 4);
    float*  Vf   = (float*) alloc((size_t)BT * Dz * 4);
    float*  XSf  = (float*) alloc((size_t)BT * Dz * 4);
    hfloat* Kh   = (hfloat*)alloc((size_t)BT * Dz * 2);        // reused as Hh
    hfloat* Vh   = (hfloat*)alloc((size_t)BT * Dz * 2);        // reused as G1h
    float*  CS   = (float*) alloc((size_t)Bz * NC * Dz * Dz * 4);
    float*  Pfx  = (float*) alloc((size_t)Bz * NC * Dz * Dz * 4);
    float*  NUM  = (float*) alloc((size_t)BT * Dz * 4);
    float*  F1   = PRJ;                                         // dead after split_phi
    float*  F2   = PRJ + (size_t)BT * Dz;
    hfloat* Hh   = Kh;                                          // dead after chunksum
    hfloat* G1h  = Vh;                                          // dead after chunksum
    (void)ws_size; (void)n_in; (void)in_sizes; (void)out_size;

    // 1. weight convert/transpose + 2. LayerNorm
    wcvt_kernel<<<(512 * 512 + 2 * Dz * Dz) / 256, 256, 0, stream>>>(Wk, Wq, Wv, sc_w, w1, w2,
                                                                     WtP, W1t, W2t);
    ln_kernel<<<BT, 256, 0, stream>>>(hist, ln_g, ln_b, Xh);
    // 3. fused projection GEMM: PRJ = Xh @ [Wk|Wq|Wv|sc_w]  (M=8192,N=512,K=512)
    gemm_wmma_kernel<<<dim3(512 / 16, BT / 16), 32, 0, stream>>>(Xh, WtP, PRJ, 512, 512);
    // 4. split + phi
    split_phi_kernel<<<(BT * 512) / 256, 256, 0, stream>>>(PRJ, Kf, Qf, Vf, XSf, Kh, Vh);
    // 5. Z cumsum -> output region 2
    zcum_kernel<<<2, 256, 0, stream>>>(Kf, Z0, Zout);
    // 6. per-chunk KV sums (WMMA) + 7. exclusive prefix
    chunksum_kernel<<<dim3(Dz / 16, Dz / 16, Bz * NC), 32, 0, stream>>>(Kh, Vh, CS);
    prefix_kernel<<<(Bz * Dz * Dz) / 256, 256, 0, stream>>>(CS, Pfx);
    // 8. fused S-scan + numerator -> output region 1
    scan_kernel<<<Bz * NC, 1024, 0, stream>>>(Kf, Vf, Qf, Pfx, S0, Sout, NUM);
    // 9. denominator + attention output (f16)
    den_attn_kernel<<<BT / 256, 256, 0, stream>>>(Qf, Zout, NUM, Hh);
    // 10-13. FFN: two WMMA GEMMs + epilogues, residual -> output region 0
    gemm_wmma_kernel<<<dim3(Dz / 16, BT / 16), 32, 0, stream>>>(Hh, W1t, F1, Dz, Dz);
    bias_relu_kernel<<<(BT * Dz) / 256, 256, 0, stream>>>(F1, b1, G1h);
    gemm_wmma_kernel<<<dim3(Dz / 16, BT / 16), 32, 0, stream>>>(G1h, W2t, F2, Dz, Dz);
    final_kernel<<<(BT * Dz) / 256, 256, 0, stream>>>(F2, b2, XSf, sc_b, Out0);
}
